// GINNodeEmbedding_49675591745679
// MI455X (gfx1250) — matrix operand order repaired and checked by reading, not compiled, not run on hardware
//
#include <hip/hip_runtime.h>
#include <stdint.h>

// ---------------------------------------------------------------------------
// GIN node embedding pipeline for MI455X (gfx1250, wave32, WMMA bf16).
// All dense algebra runs through v_wmma_f32_16x16x32_bf16.
// ---------------------------------------------------------------------------

typedef __bf16 bf16;
typedef __attribute__((ext_vector_type(16))) __bf16 v16bf;
typedef __attribute__((ext_vector_type(8)))  float  v8f;

union Frag { v16bf v; uint4 q[2]; };

// RBF tables (module constants in the reference, not passed as inputs)
__constant__ float c_bfStart[8] = {0.f, 0.f, 3.f, 0.f, 0.f, 0.f, 0.f, 0.f};
__constant__ float c_bfStep [8] = {0.1f, 0.05f, 0.3f, 0.05f, 0.05f, 0.05f, 0.5f, 0.05f};
__constant__ int   c_bfLen [8]  = {20, 20, 30, 20, 20, 20, 20, 20};
__constant__ float c_bfGam [8]  = {10.f, 1.f, 1.f, 1.f, 1.f, 1.f, 2.f, 1.f};

// ---------------------------------------------------------------------------
// WMMA core: wave computes a 16x128 strip.  A fragments in registers,
// B (transposed, [n][K] bf16) in LDS.  KC = K/32.
// ---------------------------------------------------------------------------
template<int KC>
__device__ __forceinline__ void wmma_compute(const Frag* __restrict__ afrag,
                                             const bf16* __restrict__ lB,
                                             v8f acc[8])
{
    const int lane = threadIdx.x & 31;
    const int ml   = lane & 15;
    const int kh   = lane >> 4;
    constexpr int K = KC * 32;
#pragma unroll
    for (int nt = 0; nt < 8; ++nt) {
        const bf16* bCol = lB + (size_t)(nt * 16 + ml) * K;
#pragma unroll
        for (int kc = 0; kc < KC; ++kc) {
            Frag bfr;
            bfr.q[0] = *(const uint4*)(bCol + kc * 32 + 8 * kh);
            bfr.q[1] = *(const uint4*)(bCol + kc * 32 + 8 * kh + 16);
            acc[nt] = __builtin_amdgcn_wmma_f32_16x16x32_bf16(
                false, afrag[kc].v, false, bfr.v, (short)0, acc[nt], false, false);
        }
    }
}

// C/D layout: vgpr j -> row j + 8*(lane>=16); col = ntile*16 + lane%16
__device__ __forceinline__ void wmma_store(float* __restrict__ out, int row0, int M,
                                           const v8f acc[8],
                                           const float* __restrict__ bias,
                                           bool accumulate)
{
    const int lane = threadIdx.x & 31;
    const int wave = threadIdx.x >> 5;
    const int n0   = lane & 15;
    const int kh   = lane >> 4;
#pragma unroll
    for (int nt = 0; nt < 8; ++nt) {
        const int col = nt * 16 + n0;
        const float bb = bias ? bias[col] : 0.f;
#pragma unroll
        for (int j = 0; j < 8; ++j) {
            const int gr = row0 + wave * 16 + j + 8 * kh;
            if (gr < M) {
                const size_t idx = (size_t)gr * 128 + col;
                const float v = acc[nt][j] + bb;
                if (accumulate) out[idx] += v;
                else            out[idx]  = v;
            }
        }
    }
}

// ---------------------------------------------------------------------------
// GEMM: out[M,128] = act(A[M,128]) @ W[128,128] + bias
// BN=true: act(x) = relu(x*scale[k] + shift[k]) from batch stats.
// ---------------------------------------------------------------------------
template<bool BN>
__global__ __launch_bounds__(256)
void gemm_mlp_kernel(const float* __restrict__ A, const float* __restrict__ W,
                     const float* __restrict__ bias,
                     const float* __restrict__ stats, const float* __restrict__ gam,
                     const float* __restrict__ bet,
                     float* __restrict__ out, int M, float invM)
{
    __shared__ bf16  lB[128 * 128];
    __shared__ float sScale[128];
    __shared__ float sShift[128];

    const int tid  = threadIdx.x;
    const int row0 = blockIdx.x * 128;

    if (BN && tid < 128) {
        const float mu  = stats[tid] * invM;
        const float var = stats[128 + tid] * invM - mu * mu;
        const float sc  = gam[tid] * rsqrtf(var + 1e-5f);
        sScale[tid] = sc;
        sShift[tid] = bet[tid] - mu * sc;
    }
    // stage W transposed as bf16: lB[n*128 + k]
    for (int i = tid; i < 128 * 128; i += 256) {
        const int k = i >> 7, n = i & 127;
        lB[n * 128 + k] = (bf16)W[i];
    }
    __syncthreads();

    const int lane = tid & 31, wave = tid >> 5;
    const int ml = lane & 15, kh = lane >> 4;
    const int gr = row0 + wave * 16 + ml;

    Frag afrag[4];
#pragma unroll
    for (int kc = 0; kc < 4; ++kc) {
#pragma unroll
        for (int half = 0; half < 2; ++half) {
            const int kbase = kc * 32 + 8 * kh + 16 * half;
            float4 x0 = make_float4(0.f, 0.f, 0.f, 0.f);
            float4 x1 = make_float4(0.f, 0.f, 0.f, 0.f);
            if (gr < M) {
                x0 = *(const float4*)(A + (size_t)gr * 128 + kbase);
                x1 = *(const float4*)(A + (size_t)gr * 128 + kbase + 4);
            }
            const float xs[8] = {x0.x, x0.y, x0.z, x0.w, x1.x, x1.y, x1.z, x1.w};
#pragma unroll
            for (int e = 0; e < 8; ++e) {
                float v = xs[e];
                if (BN) {
                    v = v * sScale[kbase + e] + sShift[kbase + e];
                    v = v > 0.f ? v : 0.f;
                }
                afrag[kc].v[half * 8 + e] = (bf16)v;
            }
        }
    }

    v8f zero = {};
    v8f acc[8];
#pragma unroll
    for (int i = 0; i < 8; ++i) acc[i] = zero;
    wmma_compute<4>(afrag, lB, acc);
    wmma_store(out, row0, M, acc, bias, false);
}

// ---------------------------------------------------------------------------
// Bond float embed: out[e] += RBF(xf[e]) @ Bp  (K=256 packed features,
// bias baked into Bp row (f=0,j=31) against constant-1 feature).
// ---------------------------------------------------------------------------
__global__ __launch_bounds__(256)
void gemm_bond_kernel(const float* __restrict__ xf, const uint16_t* __restrict__ BpU,
                      float* __restrict__ out, int M)
{
    __shared__ bf16 lB[128 * 256];
    const bf16* Bp = reinterpret_cast<const bf16*>(BpU);
    const int tid  = threadIdx.x;
    const int row0 = blockIdx.x * 128;

    for (int i = tid; i < 256 * 128; i += 256) {
        const int k = i >> 7, n = i & 127;
        lB[n * 256 + k] = Bp[i];
    }
    __syncthreads();

    const int lane = tid & 31, wave = tid >> 5;
    const int ml = lane & 15, kh = lane >> 4;
    const int gr = row0 + wave * 16 + ml;

    float x8[8];
#pragma unroll
    for (int f = 0; f < 8; ++f) x8[f] = (gr < M) ? xf[(size_t)gr * 8 + f] : 0.f;

    Frag afrag[8];
#pragma unroll
    for (int kc = 0; kc < 8; ++kc) {
#pragma unroll
        for (int e = 0; e < 16; ++e) {
            const int k = kc * 32 + 8 * kh + (e < 8 ? e : e + 8);
            const int f = k >> 5, j = k & 31;
            float v = 0.f;
            if (gr < M) {
                if (j < c_bfLen[f]) {
                    const float d = x8[f] - (c_bfStart[f] + c_bfStep[f] * (float)j);
                    v = __expf(-c_bfGam[f] * d * d);
                } else if (f == 0 && j == 31) {
                    v = 1.f; // bias feature
                }
            }
            afrag[kc].v[e] = (bf16)v;
        }
    }

    v8f zero = {};
    v8f acc[8];
#pragma unroll
    for (int i = 0; i < 8; ++i) acc[i] = zero;
    wmma_compute<8>(afrag, lB, acc);
    wmma_store(out, row0, M, acc, nullptr, true); // += onto int-table embed
}

// ---------------------------------------------------------------------------
// Angle embed: out[a] = [RBF32(xa0), xa1..5, 1, 0...] @ Bp  (K=64, bias baked)
// ---------------------------------------------------------------------------
__global__ __launch_bounds__(256)
void gemm_angle_kernel(const float* __restrict__ xa, const uint16_t* __restrict__ BpU,
                       float* __restrict__ out, int M)
{
    __shared__ bf16 lB[128 * 64];
    const bf16* Bp = reinterpret_cast<const bf16*>(BpU);
    const int tid  = threadIdx.x;
    const int row0 = blockIdx.x * 128;

    for (int i = tid; i < 64 * 128; i += 256) {
        const int k = i >> 7, n = i & 127;
        lB[n * 64 + k] = Bp[i];
    }
    __syncthreads();

    const int lane = tid & 31, wave = tid >> 5;
    const int ml = lane & 15, kh = lane >> 4;
    const int gr = row0 + wave * 16 + ml;

    float x6[6];
#pragma unroll
    for (int f = 0; f < 6; ++f) x6[f] = (gr < M) ? xa[(size_t)gr * 6 + f] : 0.f;

    Frag afrag[2];
#pragma unroll
    for (int kc = 0; kc < 2; ++kc) {
#pragma unroll
        for (int e = 0; e < 16; ++e) {
            const int k = kc * 32 + 8 * kh + (e < 8 ? e : e + 8);
            float v = 0.f;
            if (gr < M) {
                if (k < 32) {
                    const float d = x6[0] - 0.1f * (float)k;
                    v = __expf(-10.f * d * d);
                } else if (k < 37) {
                    v = x6[k - 31];
                } else if (k == 37) {
                    v = 1.f; // bias feature
                }
            }
            afrag[kc].v[e] = (bf16)v;
        }
    }

    v8f zero = {};
    v8f acc[8];
#pragma unroll
    for (int i = 0; i < 8; ++i) acc[i] = zero;
    wmma_compute<2>(afrag, lB, acc);
    wmma_store(out, row0, M, acc, nullptr, false);
}

// ---------------------------------------------------------------------------
// Elementwise / gather / scatter helpers
// ---------------------------------------------------------------------------
__global__ void atom_embed_kernel(float* __restrict__ h, const int* __restrict__ xi,
                                  const float* __restrict__ tbl, int Nn)
{
    const size_t gid = (size_t)blockIdx.x * 256 + threadIdx.x;
    const int n = (int)(gid >> 7), d = (int)(gid & 127);
    if (n < Nn) {
        float s = 0.f;
#pragma unroll
        for (int f = 0; f < 9; ++f) {
            const int idx = xi[(size_t)n * 9 + f];
            s += tbl[((size_t)f * 124 + idx) * 128 + d];
        }
        h[gid] = s;
    }
}

__global__ void bond_int_embed_kernel(float* __restrict__ h, const int* __restrict__ xi,
                                      const float* __restrict__ tbl, int Ee)
{
    const size_t gid = (size_t)blockIdx.x * 256 + threadIdx.x;
    const int e = (int)(gid >> 7), d = (int)(gid & 127);
    if (e < Ee) {
        float s = 0.f;
#pragma unroll
        for (int f = 0; f < 3; ++f) {
            const int idx = xi[(size_t)e * 3 + f];
            s += tbl[((size_t)f * 12 + idx) * 128 + d];
        }
        h[gid] = s;
    }
}

__global__ void init_scaled_kernel(float* __restrict__ out, const float* __restrict__ x,
                                   const float* __restrict__ epsPtr, size_t n)
{
    const size_t gid = (size_t)blockIdx.x * 256 + threadIdx.x;
    if (gid < n) out[gid] = (1.f + epsPtr[0]) * x[gid];
}

// accum[dst[e]] += relu(x[src[e]] + eattr[e])
__global__ void scatter_msg_kernel(float* __restrict__ accum, const float* __restrict__ x,
                                   const float* __restrict__ eattr,
                                   const int* __restrict__ src, const int* __restrict__ dst,
                                   int nE)
{
    const size_t gid = (size_t)blockIdx.x * 256 + threadIdx.x;
    const int e = (int)(gid >> 7), d = (int)(gid & 127);
    if (e < nE) {
        const float m = x[(size_t)src[e] * 128 + d] + eattr[(size_t)e * 128 + d];
        if (m > 0.f) atomicAdd(&accum[(size_t)dst[e] * 128 + d], m);
    }
}

__global__ __launch_bounds__(256)
void colstats_kernel(const float* __restrict__ X, float* __restrict__ stats, int M)
{
    __shared__ float s1[256];
    __shared__ float s2[256];
    const int tid = threadIdx.x;
    const int c = tid & 127, g = tid >> 7;
    float a = 0.f, b = 0.f;
    for (int r = blockIdx.x * 2 + g; r < M; r += gridDim.x * 2) {
        const float v = X[(size_t)r * 128 + c];
        a += v; b += v * v;
    }
    s1[tid] = a; s2[tid] = b;
    __syncthreads();
    if (tid < 128) {
        atomicAdd(&stats[tid],        s1[tid] + s1[tid + 128]);
        atomicAdd(&stats[128 + tid],  s2[tid] + s2[tid + 128]);
    }
}

__global__ void bn_relu_kernel(float* __restrict__ X, const float* __restrict__ stats,
                               const float* __restrict__ g, const float* __restrict__ b,
                               int M, float invM)
{
    const size_t gid = (size_t)blockIdx.x * 256 + threadIdx.x;
    if (gid >= (size_t)M * 128) return;
    const int c = (int)(gid & 127);
    const float mu  = stats[c] * invM;
    const float var = stats[128 + c] * invM - mu * mu;
    const float v = (X[gid] - mu) * rsqrtf(var + 1e-5f) * g[c] + b[c];
    X[gid] = v > 0.f ? v : 0.f;
}

// ---------------------------------------------------------------------------
// Weight packing: bond [6][256][128] (8 feats x 32 slots; bias in (f0,j31)),
// angle [5][64][128] (32 rbf + 5 linear + bias row 37)
// ---------------------------------------------------------------------------
__global__ void pack_bond_kernel(uint16_t* __restrict__ Bp, const float* __restrict__ bfW,
                                 const float* __restrict__ bfb)
{
    const int gid = blockIdx.x * 256 + threadIdx.x;
    if (gid >= 6 * 256 * 128) return;
    const int n = gid & 127;
    const int k = (gid >> 7) & 255;
    const int lt = gid >> 15;
    const int f = k >> 5, j = k & 31;
    float v = 0.f;
    if (j < 30) v = bfW[(((size_t)lt * 8 + f) * 30 + j) * 128 + n];
    if (f == 0 && j == 31) {
        v = 0.f;
        for (int ff = 0; ff < 8; ++ff) v += bfb[((size_t)lt * 8 + ff) * 128 + n];
    }
    bf16 hv = (bf16)v;
    Bp[gid] = *reinterpret_cast<uint16_t*>(&hv);
}

__global__ void pack_angle_kernel(uint16_t* __restrict__ Bp,
                                  const float* __restrict__ Wang, const float* __restrict__ bang,
                                  const float* __restrict__ Wrest, const float* __restrict__ brest)
{
    const int gid = blockIdx.x * 256 + threadIdx.x;
    if (gid >= 5 * 64 * 128) return;
    const int n = gid & 127;
    const int k = (gid >> 7) & 63;
    const int lt = gid >> 13;
    float v = 0.f;
    if (k < 32)       v = Wang[((size_t)lt * 32 + k) * 128 + n];
    else if (k < 37)  v = Wrest[((size_t)lt * 5 + (k - 32)) * 128 + n];
    else if (k == 37) v = bang[(size_t)lt * 128 + n] + brest[(size_t)lt * 128 + n];
    bf16 hv = (bf16)v;
    Bp[gid] = *reinterpret_cast<uint16_t*>(&hv);
}

// ---------------------------------------------------------------------------
// Host orchestration
// ---------------------------------------------------------------------------
extern "C" void kernel_launch(void* const* d_in, const int* in_sizes, int n_in,
                              void* d_out, int out_size, void* d_ws, size_t ws_size,
                              hipStream_t stream)
{
    constexpr int N_ = 100000, E_ = 200000, A_ = 400000, L_ = 5;
    (void)in_sizes; (void)n_in; (void)out_size; (void)ws_size;

    const int*   x_atom  = (const int*)  d_in[0];
    const int*   eidx    = (const int*)  d_in[1];   // [2,E]
    const int*   eint    = (const int*)  d_in[2];   // [E,3]
    const int*   eiba    = (const int*)  d_in[3];   // [2,A]
    const float* xf      = (const float*)d_in[4];   // [E,8]
    const float* xa      = (const float*)d_in[5];   // [A,6]
    const float* atomTbl = (const float*)d_in[6];   // [9,124,128]
    const float* bondTbl = (const float*)d_in[7];   // [6,3,12,128]
    const float* bfW     = (const float*)d_in[8];
    const float* bfb     = (const float*)d_in[9];
    const float* baWang  = (const float*)d_in[10];
    const float* babang  = (const float*)d_in[11];
    const float* baWrest = (const float*)d_in[12];
    const float* babrest = (const float*)d_in[13];
    const float* epsA  = (const float*)d_in[14];
    const float* W1a   = (const float*)d_in[15];
    const float* b1a   = (const float*)d_in[16];
    const float* bnga  = (const float*)d_in[17];
    const float* bnba  = (const float*)d_in[18];
    const float* W2a   = (const float*)d_in[19];
    const float* b2a   = (const float*)d_in[20];
    const float* epsG  = (const float*)d_in[21];
    const float* W1g   = (const float*)d_in[22];
    const float* b1g   = (const float*)d_in[23];
    const float* bngg  = (const float*)d_in[24];
    const float* bnbg  = (const float*)d_in[25];
    const float* W2g   = (const float*)d_in[26];
    const float* b2g   = (const float*)d_in[27];
    const float* obnga  = (const float*)d_in[28];
    const float* obnba  = (const float*)d_in[29];
    const float* obngba = (const float*)d_in[30];
    const float* obnbba = (const float*)d_in[31];

    const int* src  = eidx;            const int* dst  = eidx + E_;
    const int* srcb = eiba;            const int* dstb = eiba + A_;

    // workspace carve (floats)
    float* ws = (float*)d_ws;
    size_t p = 0;
    float* hA     = ws + p; p += (size_t)N_ * 128;          // h ping
    float* hbaA   = ws + p; p += (size_t)E_ * 128;          // h_ba ping
    float* pre    = ws + p; p += (size_t)E_ * 128;          // scatter accumulator / GEMM1 input
    float* big    = ws + p; p += (size_t)A_ * 128;          // t1 (<=E rows) and cur_ang (A rows); disjoint lifetimes
    float* curhba = ws + p; p += (size_t)E_ * 128;          // fresh bond embed
    uint16_t* Bbond = (uint16_t*)(ws + p); p += (size_t)6 * 256 * 128 / 2;
    uint16_t* Bang  = (uint16_t*)(ws + p); p += (size_t)5 * 64 * 128 / 2;
    float* stats  = ws + p; p += 256;
    float* t1     = big;      // [<=E,128]
    float* curang = big;      // [A,128]

    float* hbuf[2]   = { hA,   (float*)d_out };
    float* hbabuf[2] = { hbaA, (float*)d_out + (size_t)N_ * 128 };

    const dim3 blk(256);
    auto eGrid = [](size_t cnt) { return dim3((unsigned)((cnt + 255) / 256)); };
    auto gGrid = [](int M)      { return dim3((unsigned)((M + 127) / 128)); };

    // ---- setup ----
    pack_bond_kernel <<<eGrid((size_t)6 * 256 * 128), blk, 0, stream>>>(Bbond, bfW, bfb);
    pack_angle_kernel<<<eGrid((size_t)5 * 64 * 128),  blk, 0, stream>>>(Bang, baWang, babang, baWrest, babrest);
    atom_embed_kernel<<<eGrid((size_t)N_ * 128), blk, 0, stream>>>(hA, x_atom, atomTbl, N_);
    bond_int_embed_kernel<<<eGrid((size_t)E_ * 128), blk, 0, stream>>>(hbaA, eint, bondTbl, E_);
    gemm_bond_kernel<<<gGrid(E_), blk, 0, stream>>>(xf, Bbond, hbaA, E_);

    for (int l = 0; l < L_; ++l) {
        float* hin   = hbuf[l & 1];   float* hout  = hbuf[(l + 1) & 1];
        float* bain  = hbabuf[l & 1]; float* baout = hbabuf[(l + 1) & 1];

        // ================= atom GIN conv =================
        init_scaled_kernel<<<eGrid((size_t)N_ * 128), blk, 0, stream>>>(pre, hin, epsA + l, (size_t)N_ * 128);
        scatter_msg_kernel<<<eGrid((size_t)E_ * 128), blk, 0, stream>>>(pre, hin, bain, src, dst, E_);
        hipMemsetAsync(stats, 0, 256 * sizeof(float), stream);
        gemm_mlp_kernel<false><<<gGrid(N_), blk, 0, stream>>>(
            pre, W1a + (size_t)l * 16384, b1a + (size_t)l * 128,
            nullptr, nullptr, nullptr, t1, N_, 0.f);
        colstats_kernel<<<dim3(512), blk, 0, stream>>>(t1, stats, N_);
        gemm_mlp_kernel<true><<<gGrid(N_), blk, 0, stream>>>(
            t1, W2a + (size_t)l * 16384, b2a + (size_t)l * 128,
            stats, bnga + (size_t)l * 128, bnba + (size_t)l * 128, hout, N_, 1.f / N_);
        if (l < L_ - 1) {
            hipMemsetAsync(stats, 0, 256 * sizeof(float), stream);
            colstats_kernel<<<dim3(512), blk, 0, stream>>>(hout, stats, N_);
            bn_relu_kernel<<<eGrid((size_t)N_ * 128), blk, 0, stream>>>(
                hout, stats, obnga + (size_t)l * 128, obnba + (size_t)l * 128, N_, 1.f / N_);
        }

        // ================= bond GIN conv =================
        bond_int_embed_kernel<<<eGrid((size_t)E_ * 128), blk, 0, stream>>>(
            curhba, eint, bondTbl + (size_t)(l + 1) * 3 * 12 * 128, E_);
        gemm_bond_kernel<<<gGrid(E_), blk, 0, stream>>>(
            xf, Bbond + (size_t)(l + 1) * 256 * 128, curhba, E_);
        gemm_angle_kernel<<<gGrid(A_), blk, 0, stream>>>(
            xa, Bang + (size_t)l * 64 * 128, curang, A_);
        init_scaled_kernel<<<eGrid((size_t)E_ * 128), blk, 0, stream>>>(pre, curhba, epsG + l, (size_t)E_ * 128);
        scatter_msg_kernel<<<eGrid((size_t)A_ * 128), blk, 0, stream>>>(pre, curhba, curang, srcb, dstb, A_);
        hipMemsetAsync(stats, 0, 256 * sizeof(float), stream);
        gemm_mlp_kernel<false><<<gGrid(E_), blk, 0, stream>>>(
            pre, W1g + (size_t)l * 16384, b1g + (size_t)l * 128,
            nullptr, nullptr, nullptr, t1, E_, 0.f);
        colstats_kernel<<<dim3(512), blk, 0, stream>>>(t1, stats, E_);
        gemm_mlp_kernel<true><<<gGrid(E_), blk, 0, stream>>>(
            t1, W2g + (size_t)l * 16384, b2g + (size_t)l * 128,
            stats, bngg + (size_t)l * 128, bnbg + (size_t)l * 128, baout, E_, 1.f / E_);
        if (l < L_ - 1) {
            hipMemsetAsync(stats, 0, 256 * sizeof(float), stream);
            colstats_kernel<<<dim3(512), blk, 0, stream>>>(baout, stats, E_);
            bn_relu_kernel<<<eGrid((size_t)E_ * 128), blk, 0, stream>>>(
                baout, stats, obngba + (size_t)l * 128, obnbba + (size_t)l * 128, E_, 1.f / E_);
        }
    }
}
